// SparseNeighborhoodAttentionBlock_2052994367931
// MI455X (gfx1250) — matrix operand order repaired and checked by reading, not compiled, not run on hardware
//
#include <hip/hip_runtime.h>

// ---------------------------------------------------------------------------
// Sparse neighborhood attention block, fused per-query workgroup, MI455X.
//  - K projection via v_wmma_f32_16x16x32_bf16 (A from LDS, B pre-swizzled).
//  - Fused RoPE(k) + scores on WMMA C-tiles using DPP cross-lane ops
//    (quad_perm pair exchange, row_shr add tree) instead of ds_bpermute.
//  - V projection algebraically folded: out_h = (sum_l attn[h,l] G[l]) @ Wv_h^T,
//    and the attention-weighted sum itself done with WMMA (attn as bf16 A).
//  - All weight matrices pre-converted to bf16 in workspace (halves L2 traffic).
// ---------------------------------------------------------------------------

typedef __bf16 v16bf __attribute__((ext_vector_type(16)));
typedef float  v8f   __attribute__((ext_vector_type(8)));

#define DDIM    256
#define NHEAD   8
#define HDIM    32
#define L_TOT   164
#define LPAD    176          // 11 tiles of 16 (K-projection M dimension)
#define GROWS   192          // G rows incl. zero pad (k-dim for attn*G WMMA)
#define GSTRIDE 528          // bytes per gathered row: 264 bf16 (16B aligned)
#define ASTRIDE 384          // bytes per attn row: 192 bf16
#define NT_M    11
#define KSTEPS  8            // 256 / 32
#define KSTEPS_L 6           // 192 / 32

// workspace layout (halfs)
#define WS_WKFRAG 0          // 16*8*32*16 = 65536 halfs (128KB)
#define WS_WQ     65536
#define WS_WV     131072
#define WS_WO     196608

// LDS partition (bytes); total 129024
#define OFF_G      0          // 192*528 = 101376
#define OFF_QRAW   101376
#define OFF_QR     102400
#define OFF_XN     103424
#define OFF_SC     104448     // scores: 8*176 f32 = 5632
#define OFF_ATTNB  110080     // attn bf16, 16 rows x 192 halfs = 6144
#define OFF_GH     116224     // 8*256 f32 = 8192
#define OFF_OUTV   124416     // 256 f32
#define OFF_PI     125440
#define OFF_PJ     126144
#define OFF_PL     126848
#define OFF_ROW    127552
#define OFF_OOB    128256
#define OFF_RED    128960
#define OFF_STAT   128992
#define SMEM_BYTES 129024

// DPP helpers (pure VALU cross-lane, no LDS round-trip)
#define DPP_XOR1(x)                                                        \
  __int_as_float(__builtin_amdgcn_update_dpp(__float_as_int(x),            \
      __float_as_int(x), 0xB1 /*quad_perm [1,0,3,2]*/, 0xF, 0xF, false))
#define DPP_ADD_SHR(x, ctrl)                                               \
  ((x) + __int_as_float(__builtin_amdgcn_update_dpp(0,                     \
      __float_as_int(x), (ctrl), 0xF, 0xF, true)))

__device__ __forceinline__ unsigned short f2bf(float f) {
  unsigned u = __float_as_uint(f);
  u += 0x7FFFu + ((u >> 16) & 1u);        // round-to-nearest-even
  return (unsigned short)(u >> 16);
}
__device__ __forceinline__ float bf_lo(unsigned u) { return __uint_as_float(u << 16); }
__device__ __forceinline__ float bf_hi(unsigned u) { return __uint_as_float(u & 0xFFFF0000u); }
__device__ __forceinline__ float waveSum(float v) {
  for (int m = 16; m >= 1; m >>= 1) v += __shfl_xor(v, m, 32);
  return v;
}

// dot(256) of an f32 LDS vector against a bf16 weight row (uint4 = 8 weights)
__device__ __forceinline__ float dotRowBf16(const float* x, const unsigned short* row) {
  const uint4* wr = (const uint4*)row;
  float acc = 0.f;
#pragma unroll 4
  for (int d8 = 0; d8 < DDIM / 8; ++d8) {
    uint4 w = wr[d8];
    int d = d8 * 8;
    acc += x[d + 0] * bf_lo(w.x) + x[d + 1] * bf_hi(w.x)
         + x[d + 2] * bf_lo(w.y) + x[d + 3] * bf_hi(w.y)
         + x[d + 4] * bf_lo(w.z) + x[d + 5] * bf_hi(w.z)
         + x[d + 6] * bf_lo(w.w) + x[d + 7] * bf_hi(w.w);
  }
  return acc;
}

// ---- kernel 0a: swizzle Wk (f32 row-major) into bf16 B-fragments ----------
// frag layout: [nt(16)][kk(8)][lane(32)][16 halfs]; lane L holds column
// n = nt*16 + (L&15), K = kk*32 + (L<16 ? 0..7,16..23 : 8..15,24..31).
__global__ __launch_bounds__(256) void prep_wk_frags(const float* __restrict__ Wk,
                                                     unsigned short* __restrict__ frag) {
  int flat = blockIdx.x * 256 + threadIdx.x;     // 0 .. 65535
  int th   = flat & 15;
  int lane = (flat >> 4) & 31;
  int kk   = (flat >> 9) & 7;
  int nt   = flat >> 12;
  int n = nt * 16 + (lane & 15);
  int c = kk * 32 + ((lane < 16) ? 0 : 8) + (th & 7) + ((th & 8) << 1);
  frag[flat] = f2bf(Wk[n * DDIM + c]);
}

// ---- kernel 0b: Wq/Wv/Wo f32 -> bf16 row-major ----------------------------
__global__ __launch_bounds__(256) void prep_w_bf16(const float* __restrict__ Wq,
                                                   const float* __restrict__ Wv,
                                                   const float* __restrict__ Wo,
                                                   unsigned short* __restrict__ ws) {
  int flat  = blockIdx.x * 256 + threadIdx.x;    // 0 .. 196607
  int which = flat >> 16;
  int elem  = flat & 65535;
  const float* src = (which == 0) ? Wq : (which == 1) ? Wv : Wo;
  ws[WS_WQ + flat] = f2bf(src[elem]);
}

// ---- main fused kernel: one query per workgroup, 8 wave32 -----------------
__global__ __launch_bounds__(256) void sna_main(
    const float* __restrict__ query, const float* __restrict__ qpos,
    const int*   __restrict__ qbo,   int nbo,
    const float* __restrict__ fmaps, const int* __restrict__ shapes,
    const float* __restrict__ ln_s,  const float* __restrict__ ln_b,
    const unsigned short* __restrict__ ws,       // wkfrag + bf16 Wq/Wv/Wo
    float* __restrict__ out) {
  extern __shared__ __align__(16) char smem[];
  char*  Gc      = smem + OFF_G;
  char*  attnBc  = smem + OFF_ATTNB;
  float* qrawS   = (float*)(smem + OFF_QRAW);
  float* qrS     = (float*)(smem + OFF_QR);
  float* xnS     = (float*)(smem + OFF_XN);
  float* scoresS = (float*)(smem + OFF_SC);
  float* ghS     = (float*)(smem + OFF_GH);
  float* outvS   = (float*)(smem + OFF_OUTV);
  float* posiS   = (float*)(smem + OFF_PI);
  float* posjS   = (float*)(smem + OFF_PJ);
  float* poslS   = (float*)(smem + OFF_PL);
  int*   rowoffS = (int*)  (smem + OFF_ROW);
  int*   oobS    = (int*)  (smem + OFF_OOB);
  float* redS    = (float*)(smem + OFF_RED);
  float* statS   = (float*)(smem + OFF_STAT);

  const unsigned short* wkfrag = ws + WS_WKFRAG;
  const unsigned short* wqb    = ws + WS_WQ;
  const unsigned short* wvb    = ws + WS_WV;
  const unsigned short* wob    = ws + WS_WO;

  const int t    = threadIdx.x;
  const int lane = t & 31;
  const int wave = t >> 5;
  const int q    = blockIdx.x;

  const float qpi = qpos[q * 2 + 0];
  const float qpj = qpos[q * 2 + 1];

  // ---- (a) neighbor metadata + LN partial sum --------------------------
  if (t < LPAD) {
    if (t < L_TOT) {
      int level = (t < 9) ? 0 : (t < 34) ? 1 : (t < 83) ? 2 : 3;
      int start = (level == 0) ? 0 : (level == 1) ? 9 : (level == 2) ? 34 : 83;
      int s  = 3 + 2 * level;
      int li = t - start;
      int hw = 1 + level;
      int di = li / s - hw;
      int dj = li % s - hw;
      int shi = shapes[level * 2 + 0], shj = shapes[level * 2 + 1];
      int mh = 0, mw = 0;
      for (int lv = 0; lv < 4; ++lv) {
        mh = max(mh, shapes[lv * 2 + 0]);
        mw = max(mw, shapes[lv * 2 + 1]);
      }
      int bi = (int)floorf(qpi * (float)shi / (float)mh);
      int bj = (int)floorf(qpj * (float)shj / (float)mw);
      int ii = bi + di, jj = bj + dj;
      int oob = (ii < 0) | (ii >= shi) | (jj < 0) | (jj >= shj);
      int ic = min(max(ii, 0), 63), jc = min(max(jj, 0), 63);
      int b = 0;
      for (int i = 1; i < nbo; ++i) if (q >= qbo[i]) b = i;   // searchsorted-right - 1
      rowoffS[t] = (((b * 4 + level) * 64 + ic) * 64 + jc) * DDIM;
      posiS[t] = (float)ii; posjS[t] = (float)jj; poslS[t] = (float)level;
      oobS[t] = oob;
    } else {            // pad rows
      rowoffS[t] = 0; posiS[t] = 0.f; posjS[t] = 0.f; poslS[t] = 0.f; oobS[t] = 1;
    }
  }
  const float qx = query[q * DDIM + t];
  {
    float s1 = waveSum(qx);
    if (lane == 0) redS[wave] = s1;
  }
  __syncthreads();
  if (t == 0) {
    float m = 0.f;
    for (int i = 0; i < 8; ++i) m += redS[i];
    statS[0] = m * (1.0f / DDIM);
  }
  __syncthreads();
  const float mean = statS[0];
  const float dx = qx - mean;
  {
    float s2 = waveSum(dx * dx);
    if (lane == 0) redS[wave] = s2;
  }
  __syncthreads();
  if (t == 0) {
    float v = 0.f;
    for (int i = 0; i < 8; ++i) v += redS[i];
    statS[1] = rsqrtf(v * (1.0f / DDIM) + 1e-5f);
  }
  __syncthreads();
  xnS[t] = dx * statS[1] * ln_s[t] + ln_b[t];
  for (int i = t; i < NHEAD * LPAD; i += 256) scoresS[i] = 0.f;     // zero scores
  for (int i = t; i < (16 * ASTRIDE) / 4; i += 256) ((unsigned*)attnBc)[i] = 0u;
  __syncthreads();

  // ---- (b) Q projection (bf16 weights) ---------------------------------
  qrawS[t] = dotRowBf16(xnS, wqb + (size_t)t * DDIM);
  __syncthreads();

  // ---- (c) RoPE(q) pre-scaled by 1/sqrt(HD); gather neighbors -> LDS ---
  {
    int p = (t & 31) >> 1;
    float fs = __expf(-(float)p * 0.0625f * 4.60517019f);  // 100^(-p/16)
    float ang = (qpi + qpj) * fs;                           // level pos = 0
    float cs, sn; __sincosf(ang, &sn, &cs);
    float me = qrawS[t], other = qrawS[t ^ 1];
    float rot = ((t & 1) == 0) ? (me * cs - other * sn) : (other * sn + me * cs);
    qrS[t] = rot * 0.1767766953f;                           // 1/sqrt(32)
  }
  for (int l = wave; l < GROWS; l += 8) {
    uint4 st = make_uint4(0u, 0u, 0u, 0u);
    if (l < L_TOT && !oobS[l]) {
      const float* src = fmaps + rowoffS[l] + lane * 8;
      float4 f0 = *(const float4*)src;
      float4 f1 = *(const float4*)(src + 4);
      st.x = (unsigned)f2bf(f0.x) | ((unsigned)f2bf(f0.y) << 16);
      st.y = (unsigned)f2bf(f0.z) | ((unsigned)f2bf(f0.w) << 16);
      st.z = (unsigned)f2bf(f1.x) | ((unsigned)f2bf(f1.y) << 16);
      st.w = (unsigned)f2bf(f1.z) | ((unsigned)f2bf(f1.w) << 16);
    }
    *(uint4*)(Gc + l * GSTRIDE + lane * 16) = st;
  }
  __syncthreads();

  // ---- (d) K = G @ Wk^T via WMMA bf16, fused RoPE(k) + q.k scores ------
  {
    const int rowAdd = (lane >= 16) ? 8 : 0;   // C layout: lanes 16-31 hold M=v+8
    const int aoff   = (lane < 16) ? 0 : 16;   // A layout: upper lanes hold K+8 halves
    for (int nt = wave * 2; nt <= wave * 2 + 1; ++nt) {
      const int h   = nt >> 1;
      const int col = nt * 16 + (lane & 15);
      const int p   = (col & 31) >> 1;
      const float fs  = __expf(-(float)p * 0.0625f * 4.60517019f);
      const float fl  = __expf(-(float)p * 0.0625f * 2.30258509f);
      const float sgn = (col & 1) ? 1.f : -1.f;
      const float qv  = qrS[col];
      const char* bbase = (const char*)wkfrag + (size_t)nt * 8192;  // 8 kk * 32 lanes * 32B
      for (int mt = 0; mt < NT_M; ++mt) {
        v8f acc = {0.f, 0.f, 0.f, 0.f, 0.f, 0.f, 0.f, 0.f};
#pragma unroll
        for (int kk = 0; kk < KSTEPS; ++kk) {
          union { uint4 u[2]; v16bf v; } A, B;
          const char* ap = Gc + (mt * 16 + (lane & 15)) * GSTRIDE + kk * 64 + aoff;
          A.u[0] = *(const uint4*)ap;
          A.u[1] = *(const uint4*)(ap + 32);
          const char* bp = bbase + (kk * 32 + lane) * 32;
          B.u[0] = *(const uint4*)bp;
          B.u[1] = *(const uint4*)(bp + 16);
          acc = __builtin_amdgcn_wmma_f32_16x16x32_bf16(
              false, A.v, false, B.v, (short)0, acc, false, false);
        }
#pragma unroll
        for (int v = 0; v < 8; ++v) {
          int row = mt * 16 + v + rowAdd;
          float val  = acc[v];
          float part = DPP_XOR1(val);                  // pair partner (col^1), VALU only
          float ang = (posiS[row] + posjS[row]) * fs + poslS[row] * fl;
          float cs, sn; __sincosf(ang, &sn, &cs);
          float red = (val * cs + sgn * part * sn) * qv;  // even: -sin, odd: +sin
          red = DPP_ADD_SHR(red, 0x111);               // row_shr:1
          red = DPP_ADD_SHR(red, 0x112);               // row_shr:2
          red = DPP_ADD_SHR(red, 0x114);               // row_shr:4
          red = DPP_ADD_SHR(red, 0x118);               // row_shr:8 -> lane15 has sum
          if ((lane & 15) == 15) atomicAdd(&scoresS[h * LPAD + row], red);
        }
      }
    }
  }
  __syncthreads();

  // ---- (e) per-head softmax over 164 neighbors (wave == head) ----------
  //        writes bf16 attn weights into 16x192 A-layout buffer (rows 8-15 = 0)
  {
    const int h = wave;
    float m = -1e30f;
    for (int l = lane; l < L_TOT; l += 32) {
      float s = oobS[l] ? -1e9f : scoresS[h * LPAD + l];
      scoresS[h * LPAD + l] = s;
      m = fmaxf(m, s);
    }
    for (int mm = 16; mm >= 1; mm >>= 1) m = fmaxf(m, __shfl_xor(m, mm, 32));
    float sum = 0.f;
    for (int l = lane; l < L_TOT; l += 32) {
      float e = __expf(scoresS[h * LPAD + l] - m);
      scoresS[h * LPAD + l] = e;
      sum += e;
    }
    sum = waveSum(sum);
    float inv = 1.f / sum;
    for (int l = lane; l < L_TOT; l += 32) {
      float a = scoresS[h * LPAD + l] * inv;
      *(unsigned short*)(attnBc + h * ASTRIDE + l * 2) = f2bf(a);
    }
  }
  __syncthreads();

  // ---- (f) gh = attn(8x192) x G(192x256) via WMMA ----------------------
  {
    const int aoff  = (lane < 16) ? 0 : 16;      // A: upper lanes hold K+8 halves
    const int koffB = (lane < 16) ? 0 : 8;       // B: upper lanes hold K+8 halves
    for (int ct = wave * 2; ct <= wave * 2 + 1; ++ct) {   // 16 col-tiles / 8 waves
      v8f acc = {0.f, 0.f, 0.f, 0.f, 0.f, 0.f, 0.f, 0.f};
      const int ccol = ct * 16 + (lane & 15);
#pragma unroll
      for (int kk = 0; kk < KSTEPS_L; ++kk) {
        union { uint4 u[2]; v16bf v; } A;
        const char* ap = attnBc + (lane & 15) * ASTRIDE + kk * 64 + aoff;
        A.u[0] = *(const uint4*)ap;
        A.u[1] = *(const uint4*)(ap + 32);
        union { unsigned short hh[16]; v16bf v; } B;
#pragma unroll
        for (int i = 0; i < 16; ++i) {           // column walk of row-major G
          int l = kk * 32 + koffB + (i & 7) + ((i & 8) << 1);
          B.hh[i] = *(const unsigned short*)(Gc + l * GSTRIDE + ccol * 2);
        }
        acc = __builtin_amdgcn_wmma_f32_16x16x32_bf16(
            false, A.v, false, B.v, (short)0, acc, false, false);
      }
      if (lane < 16) {                           // rows 0-7 hold heads, rest zero
#pragma unroll
        for (int v = 0; v < 8; ++v) ghS[v * DDIM + ct * 16 + (lane & 15)] = acc[v];
      }
    }
  }
  __syncthreads();

  // ---- (g) out = gh @ Wv^T (bf16 weights) ------------------------------
  outvS[t] = dotRowBf16(ghS + (t >> 5) * DDIM, wvb + (size_t)t * DDIM);
  __syncthreads();

  // ---- (h) residual + output projection Wo (bf16 weights) --------------
  out[q * DDIM + t] = qx + dotRowBf16(outvS, wob + (size_t)t * DDIM);
}

// ---------------------------------------------------------------------------
extern "C" void kernel_launch(void* const* d_in, const int* in_sizes, int n_in,
                              void* d_out, int out_size, void* d_ws, size_t ws_size,
                              hipStream_t stream) {
  const float* query  = (const float*)d_in[0];
  const float* qpos   = (const float*)d_in[1];
  const int*   qbo    = (const int*)  d_in[2];
  const float* fmaps  = (const float*)d_in[3];
  const int*   shapes = (const int*)  d_in[4];
  const float* ln_s   = (const float*)d_in[5];
  const float* ln_b   = (const float*)d_in[6];
  const float* Wq     = (const float*)d_in[7];
  const float* Wk     = (const float*)d_in[8];
  const float* Wv     = (const float*)d_in[9];
  const float* Wo     = (const float*)d_in[10];
  float* out = (float*)d_out;
  const int n_q = in_sizes[0] / DDIM;
  const int nbo = in_sizes[2];

  unsigned short* ws = (unsigned short*)d_ws;   // 512KB: Wk frags + bf16 Wq/Wv/Wo

  (void)n_in; (void)out_size; (void)ws_size;
  hipFuncSetAttribute(reinterpret_cast<const void*>(sna_main),
                      hipFuncAttributeMaxDynamicSharedMemorySize, SMEM_BYTES);

  prep_wk_frags<<<256, 256, 0, stream>>>(Wk, ws + WS_WKFRAG);
  prep_w_bf16<<<768, 256, 0, stream>>>(Wq, Wv, Wo, ws);
  sna_main<<<n_q, 256, SMEM_BYTES, stream>>>(query, qpos, qbo, nbo, fmaps, shapes,
                                             ln_s, ln_b, ws, out);
}
